// AnomalyTransformer_21277267984500
// MI455X (gfx1250) — matrix operand order repaired and checked by reading, not compiled, not run on hardware
//
#include <hip/hip_runtime.h>
#include <math.h>

typedef __attribute__((ext_vector_type(16))) _Float16 v16h;
typedef __attribute__((ext_vector_type(8)))  float    v8f;
typedef __attribute__((ext_vector_type(4)))  float    f4;
typedef __attribute__((ext_vector_type(4)))  _Float16 h4;

#define BT    128
#define TT    100
#define ENC   38
#define DM    512
#define NH    8
#define HD    64
#define NL    3
#define DFF   512
#define DVAE  128
#define DMC   64
#define DC    64
#define KCB   50            // codebook size K
#define NTOK  (BT*TT)       // 12800
#define OT_EPS 0.1f

// ---------------------------------------------------------------------------
// WMMA GEMM: C[M,N] = act(A[M,K] @ B + bias)
// CONTRACT (all call sites obey): M % 128 == 0, K % 32 == 0, N % 4 == 0.
// TB==0: B is [K,N] row-major. TB==1: B is [N,K] (torch linear weight).
// act: 0=none, 1=gelu(exact erf), 2=relu.
// Block = 256 threads = 8 waves; C tile = 128x64 (wave w: rows m0+16w..+15,
// 4 WMMA col-fragments). Register-pipelined staging: global loads for slab
// k+32 are issued between the barrier and the WMMAs of slab k. Base pointers
// are loop-invariant (indexing by k0) so loads lower as global_load_b128,
// keeping them off the DScnt path. LDS holds fragments in a contiguous-k order for
// v_wmma_f32_16x16x32_f16 (A and B match; not AMD's table order):
//   A 16-bit 16x32: lane g*16+m, half e -> K = e + 16g
//   B 16-bit 32x16: lane (k/16)*16+n,  half = k%16
// ---------------------------------------------------------------------------
template <int TB>
__global__ void __launch_bounds__(256)
gemm_wmma(const float* __restrict__ A, const float* __restrict__ Bw,
          const float* __restrict__ bias, float* __restrict__ C,
          int M, int N, int K, int act)
{
  __shared__ alignas(64) _Float16 As[8][32][16];   // 8 m-subtiles, 8KB
  __shared__ alignas(64) _Float16 Bs[4][32][16];   // 4 n-subtiles, 4KB
  const int tid  = threadIdx.x;
  const int wave = tid >> 5;
  const int lane = tid & 31;
  const int m0 = blockIdx.y * 128;
  const int n0 = blockIdx.x * 64;

  // ---- per-thread staging coordinates (loop-invariant bases) ------------
  const float* aB[4];
  _Float16*    adst[4];
#pragma unroll
  for (int i = 0; i < 4; ++i) {
    int idx = i * 256 + tid;            // 0..1023
    int r   = idx >> 3;                 // tile row 0..127
    int kk  = (idx & 7) << 2;           // 0,4,...,28
    aB[i] = A + (size_t)(m0 + r) * K + kk;
    int g  = kk >> 4;
    int e0 = kk & 15;
    adst[i] = &As[r >> 4][g * 16 + (r & 15)][e0];
  }
  const float* bB[2];
  _Float16*    bdst[2];
  float        bmask[2];
  if (TB) {
    // Bw [N,K]: float4 along K; quad -> one contiguous b64 LDS store
#pragma unroll
    for (int i = 0; i < 2; ++i) {
      int idx = i * 256 + tid;          // 0..511
      int n   = idx >> 3;               // 0..63
      int kk  = (idx & 7) << 2;
      int gn  = n0 + n;
      int gc  = gn < N ? gn : N - 1;    // clamp addr, mask value below
      bmask[i] = gn < N ? 1.0f : 0.0f;
      bB[i] = Bw + (size_t)gc * K + kk;
      bdst[i] = &Bs[n >> 4][((kk >> 4) << 4) + (n & 15)][kk & 15];
    }
  } else {
    // Bw [K,N]: float4 along N; 4 b16 stores, rows 16 halves apart
#pragma unroll
    for (int i = 0; i < 2; ++i) {
      int idx = i * 256 + tid;          // 0..511
      int kk  = idx >> 4;               // 0..31
      int n4  = (idx & 15) << 2;        // 0..60 (quad inside one subtile)
      int gn0 = n0 + n4;
      int gc  = gn0 < N ? gn0 : N - 4;  // N%4==0
      bmask[i] = gn0 < N ? 1.0f : 0.0f;
      bB[i] = Bw + (size_t)kk * N + gc;
      bdst[i] = &Bs[n4 >> 4][((kk >> 4) << 4) + (n4 & 15)][kk & 15];
    }
  }

  v8f accs[4] = {{}, {}, {}, {}};
  f4 ra[4], rb[2];

  // prologue: load first slab into registers
#pragma unroll
  for (int i = 0; i < 4; ++i) ra[i] = *(const f4*)(aB[i]);
#pragma unroll
  for (int i = 0; i < 2; ++i) rb[i] = *(const f4*)(bB[i]);

  for (int k0 = 0; k0 < K; k0 += 32) {
    // ---- drain staged registers to LDS (f32 -> f16) ---------------------
#pragma unroll
    for (int i = 0; i < 4; ++i) {
      h4 hv = { (_Float16)ra[i].x, (_Float16)ra[i].y,
                (_Float16)ra[i].z, (_Float16)ra[i].w };
      *(h4*)adst[i] = hv;
    }
#pragma unroll
    for (int i = 0; i < 2; ++i) {
      f4 vb = rb[i] * bmask[i];
      if (TB) {
        h4 hv = { (_Float16)vb.x, (_Float16)vb.y,
                  (_Float16)vb.z, (_Float16)vb.w };
        *(h4*)bdst[i] = hv;
      } else {
        _Float16* p = bdst[i];
        p[0]  = (_Float16)vb.x;   // consecutive n -> rows 16 halves apart
        p[16] = (_Float16)vb.y;
        p[32] = (_Float16)vb.z;
        p[48] = (_Float16)vb.w;
      }
    }
    __syncthreads();
    // ---- issue next-slab global loads (overlap with WMMA below) ---------
    if (k0 + 32 < K) {
      const size_t ao = (size_t)(k0 + 32);
      const size_t bo = TB ? ao : ao * (size_t)N;
#pragma unroll
      for (int i = 0; i < 4; ++i) ra[i] = *(const f4*)(aB[i] + ao);
#pragma unroll
      for (int i = 0; i < 2; ++i) rb[i] = *(const f4*)(bB[i] + bo);
      __builtin_prefetch(aB[0] + ao + 32, 0, 1);   // speculative, next-next slab
    }
    // ---- fragments + 4 back-to-back WMMAs -------------------------------
    v16h af  = *(const v16h*)(&As[wave][lane][0]);
    v16h bf0 = *(const v16h*)(&Bs[0][lane][0]);
    v16h bf1 = *(const v16h*)(&Bs[1][lane][0]);
    v16h bf2 = *(const v16h*)(&Bs[2][lane][0]);
    v16h bf3 = *(const v16h*)(&Bs[3][lane][0]);
    accs[0] = __builtin_amdgcn_wmma_f32_16x16x32_f16(false, af, false, bf0,
                                                     (short)0, accs[0], false, false);
    accs[1] = __builtin_amdgcn_wmma_f32_16x16x32_f16(false, af, false, bf1,
                                                     (short)0, accs[1], false, false);
    accs[2] = __builtin_amdgcn_wmma_f32_16x16x32_f16(false, af, false, bf2,
                                                     (short)0, accs[2], false, false);
    accs[3] = __builtin_amdgcn_wmma_f32_16x16x32_f16(false, af, false, bf3,
                                                     (short)0, accs[3], false, false);
    __syncthreads();
  }

  // C/D layout: lane l, vreg r -> row = r + 8*(l/16), col = l%16
  int rb2 = m0 + wave * 16 + ((lane >> 4) << 3);
#pragma unroll
  for (int nt = 0; nt < 4; ++nt) {
    int col = n0 + nt * 16 + (lane & 15);
    if (col < N) {
      float bv = bias ? bias[col] : 0.0f;
#pragma unroll
      for (int r = 0; r < 8; ++r) {
        float v = accs[nt][r] + bv;
        if (act == 1)      v = 0.5f * v * (1.0f + erff(v * 0.70710678118654752f));
        else if (act == 2) v = fmaxf(v, 0.0f);
        C[(size_t)(rb2 + r) * N + col] = v;
      }
    }
  }
}

// ---------------------------------------------------------------------------
// Fused attention per (b,h): stage K,V once in LDS (padded, conflict-free),
// waves own score rows, wave32-shuffle softmax, series row -> d_out, then
// O = S @ V from LDS. Eliminates the O(T) reread of the K/V panels.
// LDS: 26 + 26 + 2 + 3.2 KB  (< 64KB static limit)
// ---------------------------------------------------------------------------
__global__ void __launch_bounds__(256)
attn_fused(const float* __restrict__ Q, const float* __restrict__ Km,
           const float* __restrict__ V, float* __restrict__ S,
           float* __restrict__ O, float scale)
{
  __shared__ float ks[TT][HD + 1];
  __shared__ float vs[TT][HD + 1];
  __shared__ float qs8[8][HD];
  __shared__ float prow[8][TT];
  int hh = blockIdx.x, b = blockIdx.y;
  int tid = threadIdx.x, w = tid >> 5, lane = tid & 31;

  for (int i = tid; i < TT * HD; i += 256) {
    int t = i >> 6, d = i & 63;
    size_t g = ((size_t)(b * TT + t)) * DM + hh * HD + d;
    ks[t][d] = Km[g];
    vs[t][d] = V[g];
  }
  __syncthreads();

  for (int it = 0; it < (TT + 7) / 8; ++it) {
    int row = it * 8 + w;
    bool active = row < TT;
    // cooperative: stage the 8 q rows of this iteration (512 floats)
#pragma unroll
    for (int rep = 0; rep < 2; ++rep) {
      int i  = rep * 256 + tid;
      int rr = i >> 6, d = i & 63;
      int gr = it * 8 + rr;
      if (gr < TT)
        qs8[rr][d] = Q[((size_t)(b * TT + gr)) * DM + hh * HD + d];
    }
    __syncthreads();

    float sv[4];
    float mx = -1e30f;
    if (active) {
#pragma unroll
      for (int j = 0; j < 4; ++j) {
        int col = lane + 32 * j;
        float dot = -1e30f;
        if (col < TT) {
          dot = 0.0f;
          for (int d = 0; d < HD; ++d) dot += qs8[w][d] * ks[col][d];
          dot *= scale;
        }
        sv[j] = dot;
        mx = fmaxf(mx, dot);
      }
    }
    for (int m = 16; m > 0; m >>= 1) mx = fmaxf(mx, __shfl_xor(mx, m));
    float sum = 0.0f;
    if (active) {
#pragma unroll
      for (int j = 0; j < 4; ++j) {
        int col = lane + 32 * j;
        float e = (col < TT) ? __expf(sv[j] - mx) : 0.0f;
        sv[j] = e;
        sum += e;
      }
    }
    for (int m = 16; m > 0; m >>= 1) sum += __shfl_xor(sum, m);
    if (active) {
      float inv = 1.0f / sum;
#pragma unroll
      for (int j = 0; j < 4; ++j) {
        int col = lane + 32 * j;
        if (col < TT) {
          float p = sv[j] * inv;
          prow[w][col] = p;
          S[(((size_t)(b * NH + hh)) * TT + row) * TT + col] = p;
        }
      }
      // O row = p . V   (prow/vs reads: broadcast / conflict-free)
      float o0 = 0.0f, o1 = 0.0f;
      for (int s = 0; s < TT; ++s) {
        float p = prow[w][s];
        o0 += p * vs[s][lane];
        o1 += p * vs[s][lane + 32];
      }
      size_t ob = ((size_t)(b * TT + row)) * DM + hh * HD;
      O[ob + lane]      = o0;
      O[ob + lane + 32] = o1;
    }
    __syncthreads();
  }
}

// ------------------------- pointwise / helper kernels ----------------------
__global__ void zero_f32(float* p, int n) {
  int i = blockIdx.x * blockDim.x + threadIdx.x;
  if (i < n) p[i] = 0.0f;
}

// im2col for circular conv1d(k=3): xc[row][w*38+c] = x[b, (t-1+w) mod T, c]
__global__ void build_xcat(const float* __restrict__ x, float* __restrict__ xc) {
  int idx = blockIdx.x * blockDim.x + threadIdx.x;
  if (idx >= NTOK * 128) return;
  int j   = idx & 127;
  int row = idx >> 7;
  float v = 0.0f;
  if (j < 3 * ENC) {
    int w = j / ENC, c = j % ENC;
    int b = row / TT, t = row % TT;
    int ts = t - 1 + w;
    if (ts < 0) ts += TT;
    if (ts >= TT) ts -= TT;
    v = x[((size_t)(b * TT + ts)) * ENC + c];
  }
  xc[idx] = v;
}

// pack emb_w [3,38,512] into [128,512] (zero-padded rows 114..127)
__global__ void pack_embw(const float* __restrict__ ew, float* __restrict__ wp) {
  int idx = blockIdx.x * blockDim.x + threadIdx.x;
  if (idx >= 128 * DM) return;
  int r = idx / DM, col = idx % DM;
  float v = 0.0f;
  if (r < 3 * ENC) v = ew[(size_t)r * DM + col];  // r = w*38+c already
  wp[idx] = v;
}

__global__ void add_posemb(float* h) {
  int idx = blockIdx.x * blockDim.x + threadIdx.x;
  if (idx >= NTOK * DM) return;
  int d   = idx & (DM - 1);
  int row = idx >> 9;
  int t   = row % TT;
  int i2  = (d >> 1) << 1;                           // 2*(d//2)
  float freq = __expf(-(float)i2 * (9.210340371976184f / (float)DM));
  float ang  = (float)t * freq;
  h[idx] += (d & 1) ? cosf(ang) : sinf(ang);
}

// LayerNorm (optionally fused residual add): out = LN(a [+ r]) * g + b
__global__ void __launch_bounds__(256)
ln_res(const float* __restrict__ a, const float* __restrict__ r,
       const float* __restrict__ g, const float* __restrict__ be,
       float* __restrict__ out, int rows, int cols)
{
  int wave = threadIdx.x >> 5;
  int lane = threadIdx.x & 31;
  int row  = blockIdx.x * 8 + wave;
  if (row >= rows) return;
  const float* ap = a + (size_t)row * cols;
  const float* rp = r ? r + (size_t)row * cols : nullptr;
  float sum = 0.0f, sumsq = 0.0f;
  for (int c = lane; c < cols; c += 32) {
    float x = ap[c] + (rp ? rp[c] : 0.0f);
    sum += x; sumsq += x * x;
  }
  for (int m = 16; m > 0; m >>= 1) {
    sum   += __shfl_xor(sum, m);
    sumsq += __shfl_xor(sumsq, m);
  }
  float mean = sum / (float)cols;
  float var  = sumsq / (float)cols - mean * mean;
  float inv  = rsqrtf(var + 1e-5f);
  float* op = out + (size_t)row * cols;
  for (int c = lane; c < cols; c += 32) {
    float x = ap[c] + (rp ? rp[c] : 0.0f);
    op[c] = (x - mean) * inv * g[c] + be[c];
  }
}

__global__ void concat_hb(const float* __restrict__ h1, const float* __restrict__ ma,
                          float* __restrict__ hb) {
  int idx = blockIdx.x * blockDim.x + threadIdx.x;
  if (idx >= NTOK * (DVAE + DMC)) return;
  int n = idx / (DVAE + DMC), j = idx % (DVAE + DMC);
  hb[idx] = (j < DVAE) ? h1[(size_t)n * DVAE + j] : ma[(size_t)n * DMC + (j - DVAE)];
}

// OT concept head: cost, entropic softmax assignment, emb = p@theta, loss sums
__global__ void __launch_bounds__(64)
ot_head(const float* __restrict__ pm, const float* __restrict__ theta,
        float* __restrict__ embc, float* ot_acc, float* mse_acc)
{
  __shared__ float spm[64];
  __shared__ float sp[64];
  __shared__ float red[64];
  int n = blockIdx.x, tid = threadIdx.x;
  spm[tid] = pm[(size_t)n * DC + tid];
  __syncthreads();
  float cost = 0.0f;
  if (tid < KCB) {
    for (int d = 0; d < DC; ++d) {
      float df = spm[d] - theta[tid * DC + d];
      cost += df * df;                  // == |pm|^2 - 2 pm.th + |th|^2
    }
  }
  float lg = (tid < KCB) ? (-cost / OT_EPS) : -1e30f;
  red[tid] = lg; __syncthreads();
  for (int s = 32; s > 0; s >>= 1) { if (tid < s) red[tid] = fmaxf(red[tid], red[tid + s]); __syncthreads(); }
  float mx = red[0]; __syncthreads();
  float ex = (tid < KCB) ? __expf(lg - mx) : 0.0f;
  red[tid] = ex; __syncthreads();
  for (int s = 32; s > 0; s >>= 1) { if (tid < s) red[tid] += red[tid + s]; __syncthreads(); }
  float sm = red[0]; __syncthreads();
  float p = ex / sm;
  sp[tid] = p;
  red[tid] = p * cost; __syncthreads();
  for (int s = 32; s > 0; s >>= 1) { if (tid < s) red[tid] += red[tid + s]; __syncthreads(); }
  if (tid == 0) atomicAdd(ot_acc, red[0]);
  __syncthreads();
  float e = 0.0f;
  for (int k2 = 0; k2 < KCB; ++k2) e += sp[k2] * theta[k2 * DC + tid];
  embc[(size_t)n * DC + tid] = e;
  float df = e - spm[tid];
  red[tid] = df * df; __syncthreads();
  for (int s = 32; s > 0; s >>= 1) { if (tid < s) red[tid] += red[tid + s]; __syncthreads(); }
  if (tid == 0) atomicAdd(mse_acc, red[0]);
}

__global__ void __launch_bounds__(256)
reduce_klc(const float* __restrict__ m, const float* __restrict__ lv,
           float* acc, int n)
{
  __shared__ float red[256];
  int tid = threadIdx.x;
  float s = 0.0f;
  for (int i = blockIdx.x * blockDim.x + tid; i < n; i += gridDim.x * blockDim.x) {
    float mm = m[i], l = lv[i];
    s += -0.5f * (1.0f - mm * mm + l - __expf(l));
  }
  red[tid] = s; __syncthreads();
  for (int st = 128; st > 0; st >>= 1) { if (tid < st) red[tid] += red[tid + st]; __syncthreads(); }
  if (tid == 0) atomicAdd(acc, red[0]);
}

__global__ void __launch_bounds__(256)
reduce_klz(const float* __restrict__ dzm, const float* __restrict__ zm,
           const float* __restrict__ zlv, const float* __restrict__ dzlv,
           float* acc, int n)
{
  __shared__ float red[256];
  int tid = threadIdx.x;
  float s = 0.0f;
  for (int i = blockIdx.x * blockDim.x + tid; i < n; i += gridDim.x * blockDim.x) {
    float dm = dzm[i] - zm[i];
    float dl = dzlv[i] - zlv[i];
    s += 0.5f * (dm * dm / __expf(zlv[i]) + __expf(dl) - 1.0f - dl);
  }
  red[tid] = s; __syncthreads();
  for (int st = 128; st > 0; st >>= 1) { if (tid < st) red[tid] += red[tid + st]; __syncthreads(); }
  if (tid == 0) atomicAdd(acc, red[0]);
}

__global__ void finalize_scalars(const float* acc, float* outp) {
  if (threadIdx.x == 0 && blockIdx.x == 0) {
    outp[0] = acc[0] / (float)NTOK;                 // otloss
    outp[1] = acc[1] / (float)(NTOK * DC);          // loss_mse
    outp[2] = acc[2] / (float)(NTOK * DMC);         // KL_c
    outp[3] = acc[3] / (float)(NTOK * DVAE);        // KL_z
  }
}

// ---------------------------------------------------------------------------
extern "C" void kernel_launch(void* const* d_in, const int* in_sizes, int n_in,
                              void* d_out, int out_size, void* d_ws, size_t ws_size,
                              hipStream_t stream)
{
  (void)in_sizes; (void)n_in; (void)out_size; (void)ws_size;
  const float* x      = (const float*)d_in[0];
  const float* emb_w  = (const float*)d_in[1];
  const float* Wq     = (const float*)d_in[2];
  const float* Wk     = (const float*)d_in[3];
  const float* Wv     = (const float*)d_in[4];
  const float* Wo     = (const float*)d_in[5];
  const float* bq     = (const float*)d_in[6];
  const float* bk     = (const float*)d_in[7];
  const float* bv     = (const float*)d_in[8];
  const float* bo     = (const float*)d_in[9];
  const float* c1w    = (const float*)d_in[10];
  const float* c1b    = (const float*)d_in[11];
  const float* c2w    = (const float*)d_in[12];
  const float* c2b    = (const float*)d_in[13];
  const float* ln1g   = (const float*)d_in[14];
  const float* ln1b   = (const float*)d_in[15];
  const float* ln2g   = (const float*)d_in[16];
  const float* ln2b   = (const float*)d_in[17];
  const float* lnfg   = (const float*)d_in[18];
  const float* lnfb   = (const float*)d_in[19];
  const float* en2de_w= (const float*)d_in[20];
  const float* en2de_b= (const float*)d_in[21];
  const float* eh1w   = (const float*)d_in[22];
  const float* eh1b   = (const float*)d_in[23];
  const float* eh2w   = (const float*)d_in[24];
  const float* eh2b   = (const float*)d_in[25];
  const float* mamw   = (const float*)d_in[26];
  const float* mamb   = (const float*)d_in[27];
  const float* malvw  = (const float*)d_in[28];
  const float* malvb  = (const float*)d_in[29];
  const float* zmw    = (const float*)d_in[30];
  const float* zmb    = (const float*)d_in[31];
  const float* zlvw   = (const float*)d_in[32];
  const float* zlvb   = (const float*)d_in[33];
  const float* dzm1w  = (const float*)d_in[34];
  const float* dzm1b  = (const float*)d_in[35];
  const float* dzm2w  = (const float*)d_in[36];
  const float* dzm2b  = (const float*)d_in[37];
  const float* dzl1w  = (const float*)d_in[38];
  const float* dzl1b  = (const float*)d_in[39];
  const float* dzl2w  = (const float*)d_in[40];
  const float* dzl2b  = (const float*)d_in[41];
  const float* dxm1w  = (const float*)d_in[42];
  const float* dxm1b  = (const float*)d_in[43];
  const float* dxm2w  = (const float*)d_in[44];
  const float* dxm2b  = (const float*)d_in[45];
  const float* otw    = (const float*)d_in[46];
  const float* otb    = (const float*)d_in[47];
  const float* theta  = (const float*)d_in[48];
  const float* projw  = (const float*)d_in[49];
  const float* projb  = (const float*)d_in[50];

  float* outp = (float*)d_out;
  const size_t OUT_SZ = (size_t)BT * TT * ENC;                 // 486400
  const size_t SER_L  = (size_t)BT * NH * TT * TT;             // per-layer series
  float* series0 = outp + OUT_SZ;
  float* scal    = outp + OUT_SZ + (size_t)NL * SER_L;         // 4 scalars

  // ---- workspace carve-out -----------------------------------------------
  float* ws = (float*)d_ws;
  size_t off = 0;
  auto alloc = [&](size_t n) { float* p = ws + off; off += n; return p; };
  float* h    = alloc((size_t)NTOK * DM);
  float* q    = alloc((size_t)NTOK * DM);
  float* kbuf = alloc((size_t)NTOK * DM);
  float* v    = alloc((size_t)NTOK * DM);
  float* t1   = alloc((size_t)NTOK * DM);
  float* t2   = alloc((size_t)NTOK * DM);
  float* xcat = alloc((size_t)NTOK * 128);
  float* wp   = alloc((size_t)128 * DM);
  float* hvae = alloc((size_t)NTOK * DVAE);
  float* h1   = alloc((size_t)NTOK * DVAE);
  float* h2   = alloc((size_t)NTOK * DVAE);
  float* ma   = alloc((size_t)NTOK * DMC);
  float* malv = alloc((size_t)NTOK * DMC);
  float* hb   = alloc((size_t)NTOK * (DVAE + DMC));
  float* zmean= alloc((size_t)NTOK * DVAE);
  float* zlv  = alloc((size_t)NTOK * DVAE);
  float* pm   = alloc((size_t)NTOK * DC);
  float* embc = alloc((size_t)NTOK * DC);
  float* tmpd = alloc((size_t)NTOK * DVAE);
  float* dzm  = alloc((size_t)NTOK * DVAE);
  float* dzlv = alloc((size_t)NTOK * DVAE);
  float* dxm  = alloc((size_t)NTOK * DVAE);
  float* acc  = alloc(8);

  auto gemm = [&](const float* A, const float* B, const float* bias, float* C,
                  int M, int N, int K, int tB, int act) {
    dim3 g((N + 63) / 64, M / 128);     // contract: M % 128 == 0
    if (tB) gemm_wmma<1><<<g, 256, 0, stream>>>(A, B, bias, C, M, N, K, act);
    else    gemm_wmma<0><<<g, 256, 0, stream>>>(A, B, bias, C, M, N, K, act);
  };

  // 0) zero scalar accumulators
  zero_f32<<<1, 32, 0, stream>>>(acc, 8);

  // 1) embedding: circular conv as im2col + WMMA GEMM, then positional emb
  build_xcat<<<(NTOK * 128 + 255) / 256, 256, 0, stream>>>(x, xcat);
  pack_embw<<<(128 * DM + 255) / 256, 256, 0, stream>>>(emb_w, wp);
  gemm(xcat, wp, nullptr, h, NTOK, DM, 128, 0, 0);
  add_posemb<<<(NTOK * DM + 255) / 256, 256, 0, stream>>>(h);

  // 2) encoder layers
  const float scale = 0.125f;  // 1/sqrt(64)
  for (int l = 0; l < NL; ++l) {
    const float* Wql = Wq + (size_t)l * DM * DM;
    const float* Wkl = Wk + (size_t)l * DM * DM;
    const float* Wvl = Wv + (size_t)l * DM * DM;
    const float* Wol = Wo + (size_t)l * DM * DM;
    gemm(h, Wql, bq + (size_t)l * DM, q,    NTOK, DM, DM, 0, 0);
    gemm(h, Wkl, bk + (size_t)l * DM, kbuf, NTOK, DM, DM, 0, 0);
    gemm(h, Wvl, bv + (size_t)l * DM, v,    NTOK, DM, DM, 0, 0);

    float* ser = series0 + (size_t)l * SER_L;
    attn_fused<<<dim3(NH, BT), 256, 0, stream>>>(q, kbuf, v, ser, t1, scale);

    gemm(t1, Wol, bo + (size_t)l * DM, t2, NTOK, DM, DM, 0, 0);
    ln_res<<<(NTOK + 7) / 8, 256, 0, stream>>>(h, t2, ln1g + (size_t)l * DM,
                                               ln1b + (size_t)l * DM, h, NTOK, DM);
    // FFN: conv1d(k=1) pair == linear with torch [out,in] weights (transB)
    gemm(h,  c1w + (size_t)l * DFF * DM, c1b + (size_t)l * DFF, t1, NTOK, DFF, DM, 1, 1);
    gemm(t1, c2w + (size_t)l * DM * DFF, c2b + (size_t)l * DM,  t2, NTOK, DM, DFF, 1, 0);
    ln_res<<<(NTOK + 7) / 8, 256, 0, stream>>>(h, t2, ln2g + (size_t)l * DM,
                                               ln2b + (size_t)l * DM, h, NTOK, DM);
  }

  // 3) final norm + en2de
  ln_res<<<(NTOK + 7) / 8, 256, 0, stream>>>(h, nullptr, lnfg, lnfb, h, NTOK, DM);
  gemm(h, en2de_w, en2de_b, hvae, NTOK, DVAE, DM, 0, 0);

  // 4) VAE decoder (eval: z = mean, ma = ma_mean)
  gemm(hvae, eh1w, eh1b, h1, NTOK, DVAE, DVAE, 0, 0);
  gemm(h1,  eh2w,  eh2b, h2, NTOK, DVAE, DVAE, 0, 0);
  gemm(h2,  mamw,  mamb, ma,   NTOK, DMC, DVAE, 0, 0);
  gemm(h2,  malvw, malvb, malv, NTOK, DMC, DVAE, 0, 0);
  concat_hb<<<(NTOK * (DVAE + DMC) + 255) / 256, 256, 0, stream>>>(h1, ma, hb);
  gemm(hb, zmw,  zmb,  zmean, NTOK, DVAE, DVAE + DMC, 0, 0);
  gemm(hb, zlvw, zlvb, zlv,   NTOK, DVAE, DVAE + DMC, 0, 0);

  // 5) OT concept head
  gemm(ma, otw, otb, pm, NTOK, DC, DMC, 0, 0);
  ot_head<<<NTOK, 64, 0, stream>>>(pm, theta, embc, acc + 0, acc + 1);

  // 6) prior / recon MLPs
  gemm(embc, dzm1w, dzm1b, tmpd, NTOK, DVAE, DC, 0, 2);
  gemm(tmpd, dzm2w, dzm2b, dzm,  NTOK, DVAE, DVAE, 0, 0);
  gemm(embc, dzl1w, dzl1b, tmpd, NTOK, DVAE, DC, 0, 2);
  gemm(tmpd, dzl2w, dzl2b, dzlv, NTOK, DVAE, DVAE, 0, 0);
  gemm(zmean, dxm1w, dxm1b, tmpd, NTOK, DVAE, DVAE, 0, 2);
  gemm(tmpd,  dxm2w, dxm2b, dxm,  NTOK, DVAE, DVAE, 0, 0);

  // 7) KL reductions
  reduce_klc<<<512, 256, 0, stream>>>(ma, malv, acc + 2, NTOK * DMC);
  reduce_klz<<<512, 256, 0, stream>>>(dzm, zmean, zlv, dzlv, acc + 3, NTOK * DVAE);

  // 8) projector: dxm viewed as [B, T*DVAE] @ projw[12800,3800] -> out
  gemm(dxm, projw, projb, outp, BT, TT * ENC, TT * DVAE, 0, 0);

  // 9) scalars
  finalize_scalars<<<1, 1, 0, stream>>>(acc, scal);
}
